// YOLOXHead_5669356831589
// MI455X (gfx1250) — matrix operand order repaired and checked by reading, not compiled, not run on hardware
//
#include <hip/hip_runtime.h>
#include <hip/hip_fp16.h>
#include <math.h>

typedef __attribute__((ext_vector_type(16))) _Float16 v16h;
typedef __attribute__((ext_vector_type(8)))  float    v8f;

#define NCLS   80
#define TOTPOS 21504
#define BN_EPS 1e-5f

// ---------------------------------------------------------------------------
// Fold BN into conv weights: A16[m][kidx*Cin + c] = w[m][c][ky][kx] * g*rsqrt(v+eps)
// bias[m] = b[m] - mean[m]*scale.  K-chunk t (32 wide) = one (ky,kx) tap, 32 chans.
// ---------------------------------------------------------------------------
__global__ __launch_bounds__(256)
void prep_cbn(const float* __restrict__ w, const float* __restrict__ g,
              const float* __restrict__ b, const float* __restrict__ mean,
              const float* __restrict__ var,
              _Float16* __restrict__ A, float* __restrict__ bias,
              int Cin, int ks) {
  const int K = ks * ks * Cin;
  int idx = blockIdx.x * 256 + threadIdx.x;
  if (idx >= 64 * K) return;
  int mo = idx / K, k = idx % K;
  int kidx = k / Cin, c = k % Cin;
  int ky = kidx / ks, kx = kidx % ks;
  float scale = g[mo] * rsqrtf(var[mo] + BN_EPS);
  A[idx] = (_Float16)(w[((mo * Cin + c) * ks + ky) * ks + kx] * scale);
  if (k == 0) bias[mo] = b[mo] - mean[mo] * scale;
}

// ---------------------------------------------------------------------------
// Shared WMMA fragment conventions (per CDNA5 ISA 16-bit layouts):
//  A 16x32 f16 : lane L -> M=L%16; halves 0-7: K=8g+h, 8-15: K=16+8g+(h-8),
//                g = L/16  => two 16B contiguous global reads from a K-major row.
//  B 32x16 f16 : lane L -> N=L%16, K same as A  => two 16B LDS reads.
//  C/D 16x16 f32: lane L, VGPR r -> M = 8*(L/16)+r, N = L%16.
// Block: 256 thr = 8 waves; output tile 64 och x (16w x 8h) positions.
// Wave: mt = wave&3 (M tile), sg = wave>>2 (4 rows of 16 positions).
// ---------------------------------------------------------------------------

// ---- 1x1 conv (stem): K = Cin (64..256), f32 input converted while staging.
template <typename Tin>
__global__ __launch_bounds__(256)
void conv1x1_wmma(const Tin* __restrict__ in, const _Float16* __restrict__ A,
                  const float* __restrict__ bias, _Float16* __restrict__ out,
                  int Cin, int H, int W) {
  __shared__ alignas(16) _Float16 Bp[128 * 40];   // [n][k] k-stride 40 halves

  const int tid  = threadIdx.x;
  const int lane = tid & 31;
  const int wave = tid >> 5;
  const int mt   = wave & 3;
  const int sg   = wave >> 2;
  const int w0   = blockIdx.x * 16;
  const int h0   = blockIdx.y * 8;
  const int b    = blockIdx.z;
  const int K    = Cin;
  const int T    = Cin >> 5;

  const int cS  = tid & 31;     // staging channel within chunk
  const int syS = tid >> 5;     // staging row

  const int gA   = lane >> 4;
  const int mrow = mt * 16 + (lane & 15);
  const _Float16* Arow = A + (size_t)mrow * K;

  v8f acc[4] = {};

  for (int t = 0; t < T; ++t) {
    const int c0 = t << 5;

    __syncthreads();
    {
      const Tin* src = in + ((size_t)(b * Cin + c0 + cS) * H + (h0 + syS)) * W + w0;
      _Float16* dst  = Bp + (syS * 16) * 40 + cS;
#pragma unroll
      for (int sx = 0; sx < 16; ++sx) dst[sx * 40] = (_Float16)(float)src[sx];
    }
    __syncthreads();

    union { v16h v; uint4 q[2]; } af;
    const _Float16* ap = Arow + t * 32 + 8 * gA;
    af.q[0] = *(const uint4*)(ap);
    af.q[1] = *(const uint4*)(ap + 16);
    if (t + 1 < T) __builtin_prefetch(Arow + (t + 1) * 32, 0, 0);

#pragma unroll
    for (int j = 0; j < 4; ++j) {
      const int n = (sg * 4 + j) * 16 + (lane & 15);
      union { v16h v; uint4 q[2]; } bf;
      const _Float16* bp = Bp + n * 40 + 8 * gA;
      bf.q[0] = *(const uint4*)(bp);
      bf.q[1] = *(const uint4*)(bp + 16);
      acc[j] = __builtin_amdgcn_wmma_f32_16x16x32_f16(
          false, af.v, false, bf.v, (short)0, acc[j], false, false);
    }
  }

  const int wcol  = w0 + (lane & 15);
  const int mbase = mt * 16 + ((lane >> 4) << 3);
#pragma unroll
  for (int j = 0; j < 4; ++j) {
    const int h = h0 + sg * 4 + j;
#pragma unroll
    for (int r = 0; r < 8; ++r) {
      const int och = mbase + r;
      float v = acc[j][r] + bias[och];
      float s = v / (1.f + __expf(-v));
      out[(((size_t)b * 64 + och) * H + h) * W + wcol] = (_Float16)s;
    }
  }
}

// ---- 3x3 conv, Cin=Cout=64: halo tile staged ONCE (channel-contiguous),
// zero barriers in the K loop; every input element read once per block.
// tile[row][col][c] with c padded 64->72: fragment reads are 16B-aligned,
// two 16B reads 32B apart, lane stride 144B = 36 dwords -> 16 distinct LDS banks.
__global__ __launch_bounds__(256)
void conv3x3_wmma(const _Float16* __restrict__ in, const _Float16* __restrict__ A,
                  const float* __restrict__ bias, _Float16* __restrict__ out,
                  int H, int W) {
  __shared__ alignas(16) _Float16 tile[10 * 18 * 72];   // 25.9 KB

  const int tid  = threadIdx.x;
  const int lane = tid & 31;
  const int wave = tid >> 5;
  const int mt   = wave & 3;
  const int sg   = wave >> 2;
  const int w0   = blockIdx.x * 16;
  const int h0   = blockIdx.y * 8;
  const int b    = blockIdx.z;
  const int K    = 576;

  // ---- stage 10x18 halo x 64ch once: thread = (c = tid&63, row = tid>>6 step 4)
  {
    const int c  = tid & 63;
    const int r0 = tid >> 6;
    const _Float16* src = in + ((size_t)(b * 64 + c) * H) * W;
    for (int row = r0; row < 10; row += 4) {
      const int gr = h0 + row - 1;
      const bool rowok = (unsigned)gr < (unsigned)H;
      _Float16* dst = tile + (row * 18) * 72 + c;
#pragma unroll
      for (int col = 0; col < 18; ++col) {
        const int gc = w0 + col - 1;
        _Float16 v = (_Float16)0.f;
        if (rowok && (unsigned)gc < (unsigned)W) v = src[gr * W + gc];
        dst[col * 72] = v;
      }
    }
  }
  __syncthreads();

  const int gA   = lane >> 4;
  const int mrow = mt * 16 + (lane & 15);
  const _Float16* Arow = A + (size_t)mrow * K;

  v8f acc[4] = {};

  for (int t = 0; t < 18; ++t) {
    const int kidx = t >> 1;          // tap index 0..8
    const int c0   = (t & 1) << 5;    // channel block 0/32
    const int ky   = kidx / 3;
    const int kx   = kidx % 3;

    union { v16h v; uint4 q[2]; } af;
    const _Float16* ap = Arow + t * 32 + 8 * gA;
    af.q[0] = *(const uint4*)(ap);
    af.q[1] = *(const uint4*)(ap + 16);
    if (t + 1 < 18) __builtin_prefetch(Arow + (t + 1) * 32, 0, 0);

#pragma unroll
    for (int j = 0; j < 4; ++j) {
      const int sy  = sg * 4 + j;
      const int pos = (sy + ky) * 18 + (lane & 15) + kx;
      union { v16h v; uint4 q[2]; } bf;
      const _Float16* bp = tile + pos * 72 + c0 + 8 * gA;
      bf.q[0] = *(const uint4*)(bp);
      bf.q[1] = *(const uint4*)(bp + 16);
      acc[j] = __builtin_amdgcn_wmma_f32_16x16x32_f16(
          false, af.v, false, bf.v, (short)0, acc[j], false, false);
    }
  }

  const int wcol  = w0 + (lane & 15);
  const int mbase = mt * 16 + ((lane >> 4) << 3);
#pragma unroll
  for (int j = 0; j < 4; ++j) {
    const int h = h0 + sg * 4 + j;
#pragma unroll
    for (int r = 0; r < 8; ++r) {
      const int och = mbase + r;
      float v = acc[j][r] + bias[och];
      float s = v / (1.f + __expf(-v));
      out[(((size_t)b * 64 + och) * H + h) * W + wcol] = (_Float16)s;
    }
  }
}

// ---------------------------------------------------------------------------
// Prediction 1x1 convs (85x64 GEMV/pos) + decode, channel-last output.
// ---------------------------------------------------------------------------
__global__ __launch_bounds__(256)
void head_epilogue(const _Float16* __restrict__ clsf, const _Float16* __restrict__ regf,
                   const float* __restrict__ cw, const float* __restrict__ cb,
                   const float* __restrict__ rw, const float* __restrict__ rb,
                   const float* __restrict__ ow, const float* __restrict__ ob,
                   float* __restrict__ outp, int H, int W, int Bn, float stride,
                   int posoff) {
  __shared__ float sW[85 * 64];
  __shared__ float sB[85];
  for (int i = threadIdx.x; i < 80 * 64; i += 256) sW[i] = cw[i];
  for (int i = threadIdx.x; i < 4 * 64;  i += 256) sW[80 * 64 + i] = rw[i];
  for (int i = threadIdx.x; i < 64;      i += 256) sW[84 * 64 + i] = ow[i];
  for (int i = threadIdx.x; i < 80;      i += 256) sB[i] = cb[i];
  if (threadIdx.x < 4) sB[80 + threadIdx.x] = rb[threadIdx.x];
  if (threadIdx.x == 0) sB[84] = ob[0];
  __syncthreads();

  const int HW = H * W;
  int p = blockIdx.x * 256 + threadIdx.x;
  if (p >= Bn * HW) return;
  const int b = p / HW, rem = p % HW;
  const int h = rem / W, w = rem % W;

  float cf[64], rf[64];
  const _Float16* cptr = clsf + (size_t)b * 64 * HW + rem;
  const _Float16* rptr = regf + (size_t)b * 64 * HW + rem;
#pragma unroll
  for (int c = 0; c < 64; ++c) { cf[c] = (float)cptr[(size_t)c * HW];
                                 rf[c] = (float)rptr[(size_t)c * HW]; }

  float reg4[4];
#pragma unroll
  for (int j = 0; j < 4; ++j) {
    float a = sB[80 + j];
    const float* wr = sW + (80 + j) * 64;
#pragma unroll
    for (int c = 0; c < 64; ++c) a += wr[c] * rf[c];
    reg4[j] = a;
  }
  float obj = sB[84];
  {
    const float* wr = sW + 84 * 64;
#pragma unroll
    for (int c = 0; c < 64; ++c) obj += wr[c] * rf[c];
  }
  obj = 1.f / (1.f + __expf(-obj));

  float* o = outp + ((size_t)b * TOTPOS + posoff + rem) * 85;
  o[0] = (reg4[0] + (float)w) * stride;
  o[1] = (reg4[1] + (float)h) * stride;
  o[2] = __expf(reg4[2]) * stride;
  o[3] = __expf(reg4[3]) * stride;
  o[4] = obj;
  for (int cc = 0; cc < 80; ++cc) {
    float a = sB[cc];
    const float* wr = sW + cc * 64;
#pragma unroll
    for (int c = 0; c < 64; ++c) a += wr[c] * cf[c];
    o[5 + cc] = 1.f / (1.f + __expf(-a));
  }
}

// ---------------------------------------------------------------------------
extern "C" void kernel_launch(void* const* d_in, const int* in_sizes, int n_in,
                              void* d_out, int out_size, void* d_ws, size_t ws_size,
                              hipStream_t stream) {
  (void)n_in; (void)out_size; (void)ws_size;

  const int CinL[3]   = {64, 128, 256};
  const int Hs[3]     = {128, 64, 32};
  const float strf[3] = {8.f, 16.f, 32.f};
  const int posoff[3] = {0, 16384, 20480};

  // Input index maps: scheme A = dict insertion order, scheme B = jax sorted pytree.
  const bool schemeA = (in_sizes[0] == 32 * 64 * 128 * 128);
  int xi[3];
  int cbn[3][5][5];   // [lvl][layer: stem,cls1,cls2,reg1,reg2][w,g,b,m,v]
  int prd[3][3][2];   // [lvl][clsP,regP,objP][w,b]
  for (int l = 0; l < 3; ++l) {
    if (schemeA) {
      xi[l] = l;
      int base = 3 + 31 * l;
      const int lay[5] = {0, 5, 10, 15, 20};
      for (int j = 0; j < 5; ++j)
        for (int f = 0; f < 5; ++f) cbn[l][j][f] = base + lay[j] + f;
      prd[l][0][0] = base + 25; prd[l][0][1] = base + 26;
      prd[l][1][0] = base + 27; prd[l][1][1] = base + 28;
      prd[l][2][0] = base + 29; prd[l][2][1] = base + 30;
    } else {
      // sorted keys: cls1{b,g,m,v,w} cls2 cls_pred{b,w} obj_pred{b,w}
      //              reg1 reg2 reg_pred{b,w} stem{b,g,m,v,w}; x0..x2 after params
      xi[l] = 93 + l;
      int base = 31 * l;
      auto setc = [&](int layer, int off) {
        cbn[l][layer][0] = base + off + 4;  // w
        cbn[l][layer][1] = base + off + 1;  // g
        cbn[l][layer][2] = base + off + 0;  // b
        cbn[l][layer][3] = base + off + 2;  // m
        cbn[l][layer][4] = base + off + 3;  // v
      };
      setc(1, 0);   // cls1
      setc(2, 5);   // cls2
      prd[l][0][1] = base + 10; prd[l][0][0] = base + 11;  // cls_pred b,w
      prd[l][2][1] = base + 12; prd[l][2][0] = base + 13;  // obj_pred b,w
      setc(3, 14);  // reg1
      setc(4, 19);  // reg2
      prd[l][1][1] = base + 24; prd[l][1][0] = base + 25;  // reg_pred b,w
      setc(0, 26);  // stem
    }
  }

  // ---- workspace layout
  char* ws = (char*)d_ws;
  size_t off = 0;
  auto alloc = [&](size_t bytes) -> void* {
    off = (off + 255) & ~(size_t)255;
    void* p = ws + off;
    off += bytes;
    return p;
  };
  const size_t featB = (size_t)32 * 64 * 16384 * 2;   // level-0 sized, reused
  _Float16* stemB = (_Float16*)alloc(featB);
  _Float16* tmpB  = (_Float16*)alloc(featB);
  _Float16* clsB  = (_Float16*)alloc(featB);
  _Float16* regB  = (_Float16*)alloc(featB);
  _Float16* Astem = (_Float16*)alloc(64 * 256 * 2);
  _Float16* Ac[4];
  for (int i = 0; i < 4; ++i) Ac[i] = (_Float16*)alloc(64 * 576 * 2);
  float* biasArena = (float*)alloc(5 * 64 * sizeof(float));

  float* outp = (float*)d_out;
  auto F = [&](int i) { return (const float*)d_in[i]; };

  for (int l = 0; l < 3; ++l) {
    const int Cin = CinL[l], H = Hs[l], W = Hs[l];
    _Float16* Abuf[5] = {Astem, Ac[0], Ac[1], Ac[2], Ac[3]};
    float* bbuf[5];
    for (int j = 0; j < 5; ++j) bbuf[j] = biasArena + 64 * j;

    for (int j = 0; j < 5; ++j) {
      const int ks = (j == 0) ? 1 : 3;
      const int ci = (j == 0) ? Cin : 64;
      const int n  = 64 * ks * ks * ci;
      prep_cbn<<<(n + 255) / 256, 256, 0, stream>>>(
          F(cbn[l][j][0]), F(cbn[l][j][1]), F(cbn[l][j][2]),
          F(cbn[l][j][3]), F(cbn[l][j][4]), Abuf[j], bbuf[j], ci, ks);
    }

    dim3 grid(W / 16, H / 8, 32);
    conv1x1_wmma<float><<<grid, 256, 0, stream>>>(
        F(xi[l]), Astem, bbuf[0], stemB, Cin, H, W);
    conv3x3_wmma<<<grid, 256, 0, stream>>>(stemB, Ac[0], bbuf[1], tmpB, H, W);
    conv3x3_wmma<<<grid, 256, 0, stream>>>(tmpB,  Ac[1], bbuf[2], clsB, H, W);
    conv3x3_wmma<<<grid, 256, 0, stream>>>(stemB, Ac[2], bbuf[3], tmpB, H, W);
    conv3x3_wmma<<<grid, 256, 0, stream>>>(tmpB,  Ac[3], bbuf[4], regB, H, W);

    const int npos = 32 * H * W;
    head_epilogue<<<(npos + 255) / 256, 256, 0, stream>>>(
        clsB, regB,
        F(prd[l][0][0]), F(prd[l][0][1]),
        F(prd[l][1][0]), F(prd[l][1][1]),
        F(prd[l][2][0]), F(prd[l][2][1]),
        outp, H, W, 32, strf[l], posoff[l]);
  }
}
